// Attention_57595511439927
// MI455X (gfx1250) — matrix operand order repaired and checked by reading, not compiled, not run on hardware
//
#include <hip/hip_runtime.h>
#include <hip/hip_bf16.h>

// ---------------------------------------------------------------------------
// MI455X (gfx1250, wave32) implementation of the dual-attention block.
// All dense matmuls use v_wmma_f32_16x16x32_f16 (f16 in, f32 accumulate).
// Fragment layouts follow CDNA5 ISA 7.12.2:
//   A (16x32 f16): lane -> M=lane&15; VGPR v holds K = (v<4 ? 2v : 2v+8) + (lane<16?0:8), K+1
//   B (32x16 f16): lane -> N=lane&15; VGPR v holds K = (v<4 ? 2v : 2v+8) + (lane<16?0:8), K+1
//   C/D (16x16 f32): lane -> N=lane&15; VGPR r -> M = r + (lane<16?0:8)
// ---------------------------------------------------------------------------

typedef __attribute__((ext_vector_type(16))) _Float16 v16h;
typedef __attribute__((ext_vector_type(8)))  float    v8f;

union FragH { v16h v; unsigned int u[8]; };

__device__ __forceinline__ v16h frag_a(const _Float16* row, int lane) {
  FragH f;
  int koff = (lane < 16) ? 0 : 8;
#pragma unroll
  for (int vv = 0; vv < 8; ++vv) {
    int k2 = ((vv < 4) ? (2 * vv) : (2 * vv + 8)) + koff;
    f.u[vv] = *(const unsigned int*)(row + k2);
  }
  return f.v;
}

__device__ __forceinline__ v16h frag_b(const _Float16* row, int lane) {
  FragH f;
  int koff = (lane < 16) ? 0 : 8;
#pragma unroll
  for (int vv = 0; vv < 8; ++vv) {
    f.u[vv] = *(const unsigned int*)(row + ((vv < 4) ? (2 * vv) : (2 * vv + 8)) + koff);
  }
  return f.v;
}

// ---------------------------------------------------------------------------
// Generic batched GEMM: Out[bi][m][n] = sum_k W[m*K+k] * In[bi][k*N+n] (+Add)
// W shared across batch. M % 64 == 0, K % 32 == 0, N % 128 == 0.
// Block 256 threads = 8 waves; tile 64(M) x 128(N); each wave does 4 tiles.
// Double-buffered LDS staging (one barrier per K-step) + global prefetch two
// K-steps ahead (global_prefetch_b8).
// ---------------------------------------------------------------------------
__global__ __launch_bounds__(256) void k_gemm(
    const float* __restrict__ In, const float* __restrict__ W,
    const float* __restrict__ Add, float* __restrict__ Out,
    int M, int K, int N, long inBS, long outBS)
{
  const int bi = blockIdx.z;
  const float* in  = In + (long)bi * inBS;
  float*       out = Out + (long)bi * outBS;
  const float* add = Add ? (Add + (long)bi * outBS) : nullptr;

  const int m0 = blockIdx.y * 64;
  const int n0 = blockIdx.x * 128;

  __shared__ alignas(16) _Float16 As[2][64][34];
  __shared__ alignas(16) _Float16 Bs[2][128][34];

  const int tid  = threadIdx.x;
  const int lane = tid & 31;
  const int wave = tid >> 5;       // 0..7
  const int wm   = wave >> 1;      // M-tile row (x16)
  const int wn   = (wave & 1) * 64;

  // per-thread staging coordinates
  const int am = tid >> 2;            // A row
  const int akq = (tid & 3) * 8;      // A k-offset (8 floats)
  const int bk = tid >> 3;            // B k-row
  const int bnq = (tid & 7) * 16;     // B n-offset (16 floats)

  v8f acc[4];
#pragma unroll
  for (int i = 0; i < 4; ++i) acc[i] = v8f{};

  const int nk = K >> 5;

  { // stage K-step 0 into buffer 0
    const float* srcA = W + (long)(m0 + am) * K + akq;
#pragma unroll
    for (int j = 0; j < 8; ++j) As[0][am][akq + j] = (_Float16)srcA[j];
    const float* srcB = in + (long)bk * N + n0 + bnq;
#pragma unroll
    for (int j = 0; j < 16; ++j) Bs[0][bnq + j][bk] = (_Float16)srcB[j];
  }
  __syncthreads();

  for (int i = 0; i < nk; ++i) {
    const int cur = i & 1;
    if (i + 2 < nk) { // warm GL2 two K-steps ahead
      __builtin_prefetch(W + (long)(m0 + am) * K + (i + 2) * 32 + akq, 0, 0);
      __builtin_prefetch(in + (long)((i + 2) * 32 + bk) * N + n0 + bnq, 0, 0);
    }
    if (i + 1 < nk) { // stage next K-step into the other buffer
      const int nxt = cur ^ 1;
      const int k0 = (i + 1) * 32;
      const float* srcA = W + (long)(m0 + am) * K + k0 + akq;
#pragma unroll
      for (int j = 0; j < 8; ++j) As[nxt][am][akq + j] = (_Float16)srcA[j];
      const float* srcB = in + (long)(k0 + bk) * N + n0 + bnq;
#pragma unroll
      for (int j = 0; j < 16; ++j) Bs[nxt][bnq + j][bk] = (_Float16)srcB[j];
    }

    v16h a = frag_a(&As[cur][wm * 16 + (lane & 15)][0], lane);
#pragma unroll
    for (int nt = 0; nt < 4; ++nt) {
      v16h b = frag_b(&Bs[cur][wn + nt * 16 + (lane & 15)][0], lane);
      acc[nt] = __builtin_amdgcn_wmma_f32_16x16x32_f16(
          false, a, false, b, (short)0, acc[nt], false, false);
    }
    __syncthreads();
  }

  const int rowBase = m0 + wm * 16 + ((lane < 16) ? 0 : 8);
#pragma unroll
  for (int nt = 0; nt < 4; ++nt) {
    int col = n0 + wn + nt * 16 + (lane & 15);
#pragma unroll
    for (int r = 0; r < 8; ++r) {
      long idx = (long)(rowBase + r) * N + col;
      float v = acc[nt][r];
      if (add) v += add[idx];
      out[idx] = v;
    }
  }
}

// ---------------------------------------------------------------------------
// Depthwise 3x3, stride 1, SAME. NCHW, weights (C,1,3,3).
// ---------------------------------------------------------------------------
__global__ __launch_bounds__(256) void k_dwconv3(
    const float* __restrict__ in, const float* __restrict__ wd,
    float* __restrict__ out, int C, int H, int W, long total)
{
  long idx = (long)blockIdx.x * 256 + threadIdx.x;
  if (idx >= total) return;
  int x = (int)(idx % W);
  int y = (int)((idx / W) % H);
  long bc = idx / ((long)W * H);
  int c = (int)(bc % C);
  const float* wp = wd + c * 9;
  const float* plane = in + bc * H * W;
  float s = 0.0f;
#pragma unroll
  for (int dy = -1; dy <= 1; ++dy) {
    int yy = y + dy;
    if (yy < 0 || yy >= H) continue;
#pragma unroll
    for (int dx = -1; dx <= 1; ++dx) {
      int xx = x + dx;
      if (xx < 0 || xx >= W) continue;
      s += wp[(dy + 1) * 3 + (dx + 1)] * plane[(long)yy * W + xx];
    }
  }
  out[idx] = s;
}

// AvgPool2d(2,2): (4,192,256,256) -> (4,192,128,128)
__global__ __launch_bounds__(256) void k_avgpool2(
    const float* __restrict__ in, float* __restrict__ out, long total)
{
  long idx = (long)blockIdx.x * 256 + threadIdx.x;
  if (idx >= total) return;
  long bc = idx >> 14;               // plane (b*192+c)
  int p = (int)(idx & 16383);
  int y = p >> 7, x = p & 127;
  const float* base = in + (bc << 16) + ((long)(y << 1) << 8) + (x << 1);
  out[idx] = 0.25f * (base[0] + base[1] + base[256] + base[257]);
}

// Permutation equivalent to window_partition(256, ws=8) -> window_reverse(128):
// (4,192,256,256) -> (16,192,128,128)
__global__ __launch_bounds__(256) void k_permute_q1(
    const float* __restrict__ in, float* __restrict__ out, long total)
{
  long idx = (long)blockIdx.x * 256 + threadIdx.x;
  if (idx >= total) return;
  int Bp = (int)(idx / (192L * 16384));
  long rem = idx % (192L * 16384);
  int c = (int)(rem >> 14);
  int p = (int)(rem & 16383);
  int yp = p >> 7, xp = p & 127;
  int b = Bp >> 2, a = Bp & 3;
  int h1p = yp >> 3, iy = yp & 7;
  int w1p = xp >> 3, ix = xp & 7;
  int r = h1p >> 1, whi = h1p & 1;
  int h1 = a * 8 + r, w1 = whi * 16 + w1p;
  int y = h1 * 8 + iy, x = w1 * 8 + ix;
  out[idx] = in[((long)(b * 192 + c) << 16) + (y << 8) + x];
}

// Inverse L2 norm over hw per (b, channel): out[b*nc+c] = 1/max(||row||,1e-12)
__global__ __launch_bounds__(256) void k_rownorm(
    const float* __restrict__ X, float* __restrict__ out,
    long chanPitch, long batchPitch, int len, int nc)
{
  int c = blockIdx.x, b = blockIdx.y;
  const float* row = X + (long)b * batchPitch + (long)c * chanPitch;
  float s = 0.0f;
  for (int i = threadIdx.x; i < len; i += 256) { float v = row[i]; s += v * v; }
  __shared__ float red[256];
  red[threadIdx.x] = s;
  __syncthreads();
  for (int off = 128; off > 0; off >>= 1) {
    if (threadIdx.x < off) red[threadIdx.x] += red[threadIdx.x + off];
    __syncthreads();
  }
  if (threadIdx.x == 0) out[b * nc + c] = 1.0f / fmaxf(sqrtf(red[0]), 1e-12f);
}

__global__ __launch_bounds__(256) void k_zero(float* p, long n)
{
  long i = (long)blockIdx.x * 256 + threadIdx.x;
  if (i < n) p[i] = 0.0f;
}

// ---------------------------------------------------------------------------
// Level-0 channel-attention logits: S0[bh][c][d] += sum_n q[c][n]*k[d][n]
// over a 256-wide n chunk; WMMA over K=n. Block 128 = 4 waves (2x2 quadrants).
// ---------------------------------------------------------------------------
__global__ __launch_bounds__(128) void k_s0_partial(
    const float* __restrict__ Q,   // (4,192,65536)
    const float* __restrict__ KV,  // (4,384,65536), k = first 192 channels
    float* __restrict__ S0)        // (24,32,32)
{
  const long HW = 65536;
  int bh = blockIdx.z, b = bh / 6, h = bh % 6;
  long n0 = (long)blockIdx.x * 256;

  __shared__ alignas(16) _Float16 qs[32][258];
  __shared__ alignas(16) _Float16 ks[32][258];

  int tid = threadIdx.x, lane = tid & 31, wave = tid >> 5;

  {
    int c = tid >> 2, off = (tid & 3) * 64;
    const float* qrow = Q  + ((long)(b * 192 + h * 32 + c)) * HW + n0 + off;
    const float* krow = KV + ((long)(b * 384 + h * 32 + c)) * HW + n0 + off;
#pragma unroll
    for (int j = 0; j < 64; ++j) {
      qs[c][off + j] = (_Float16)qrow[j];
      ks[c][off + j] = (_Float16)krow[j];
    }
  }
  __syncthreads();

  int ct = wave >> 1, dt = wave & 1;
  v8f acc = v8f{};
  const _Float16* arow = &qs[ct * 16 + (lane & 15)][0];
  const _Float16* brow = &ks[dt * 16 + (lane & 15)][0];
#pragma unroll
  for (int kk = 0; kk < 256; kk += 32) {
    v16h a = frag_a(arow + kk, lane);
    v16h bf = frag_b(brow + kk, lane);
    acc = __builtin_amdgcn_wmma_f32_16x16x32_f16(
        false, a, false, bf, (short)0, acc, false, false);
  }
  int row0 = ct * 16 + ((lane < 16) ? 0 : 8);
  int col  = dt * 16 + (lane & 15);
#pragma unroll
  for (int r = 0; r < 8; ++r)
    atomicAdd(&S0[bh * 1024 + (row0 + r) * 32 + col], acc[r]);
}

// Scale by temp/(|q||k|), softmax over d, emit f16 probabilities.
__global__ __launch_bounds__(32) void k_s0_softmax(
    const float* __restrict__ S0, const float* __restrict__ qn,
    const float* __restrict__ kn, const float* __restrict__ temp0,
    _Float16* __restrict__ A0h)
{
  int bh = blockIdx.x, b = bh / 6, h = bh % 6;
  int c = threadIdx.x;
  float t0 = temp0[h];
  float qi = qn[b * 192 + h * 32 + c];
  float row[32];
  float mx = -1e30f;
#pragma unroll
  for (int d = 0; d < 32; ++d) {
    float v = S0[bh * 1024 + c * 32 + d] * t0 * qi * kn[b * 192 + h * 32 + d];
    row[d] = v;
    mx = fmaxf(mx, v);
  }
  float sum = 0.0f;
#pragma unroll
  for (int d = 0; d < 32; ++d) { row[d] = __expf(row[d] - mx); sum += row[d]; }
  float inv = 1.0f / sum;
#pragma unroll
  for (int d = 0; d < 32; ++d)
    A0h[bh * 1024 + c * 32 + d] = (_Float16)(row[d] * inv);
}

// Out0attn[b][h*32+c][n] = sum_d A0[c][d] * v[d][n]; WMMA M=32,K=32,N=128/blk.
__global__ __launch_bounds__(128) void k_attn0_av(
    const _Float16* __restrict__ A0h, const float* __restrict__ KV,
    float* __restrict__ Out)
{
  const long HW = 65536;
  int bh = blockIdx.z, b = bh / 6, h = bh % 6;
  long n0 = (long)blockIdx.x * 128;

  __shared__ alignas(16) _Float16 vT[128][34];
  __shared__ alignas(16) _Float16 a_s[32][34];

  int tid = threadIdx.x, lane = tid & 31, wave = tid >> 5;

  {
    int n = tid;
    const float* base = KV + ((long)(b * 384 + 192 + h * 32)) * HW + n0 + n;
#pragma unroll
    for (int d = 0; d < 32; ++d) vT[n][d] = (_Float16)base[(long)d * HW];
  }
  for (int e = tid; e < 1024; e += 128)
    a_s[e >> 5][e & 31] = A0h[bh * 1024 + e];
  __syncthreads();

  int mt = wave & 1;
  int ntbase = (wave >> 1) * 4;
  v16h a = frag_a(&a_s[mt * 16 + (lane & 15)][0], lane);
  int rowc0 = b * 192 + h * 32 + mt * 16 + ((lane < 16) ? 0 : 8);
#pragma unroll
  for (int t = 0; t < 4; ++t) {
    int nt = ntbase + t;
    v16h bf = frag_b(&vT[nt * 16 + (lane & 15)][0], lane);
    v8f acc = v8f{};
    acc = __builtin_amdgcn_wmma_f32_16x16x32_f16(
        false, a, false, bf, (short)0, acc, false, false);
    long col = n0 + nt * 16 + (lane & 15);
#pragma unroll
    for (int r = 0; r < 8; ++r)
      Out[(long)(rowc0 + r) * HW + col] = acc[r];
  }
}

// ---------------------------------------------------------------------------
// Level-1 shifted-window cosine attention. One block per (window, head).
// Q from (16,192,128,128) with roll(-4), KV window = win>>2 from (4,384,128,128).
// S = Q^T K (K=32, 1 WMMA/tile) -> scale/rpb/mask -> softmax -> attn x V^T.
// ---------------------------------------------------------------------------
__global__ __launch_bounds__(128) void k_attn1(
    const float* __restrict__ Qc, const float* __restrict__ KVd,
    const float* __restrict__ rpb, const float* __restrict__ temp1,
    float* __restrict__ X1)
{
  const long NT1 = 262144;
  int win = blockIdx.x, head = blockIdx.y;
  int Bp = win >> 8, wy = (win >> 4) & 15, wx = win & 15;
  int kwin = win >> 2;
  int kb = kwin >> 8, ky = (kwin >> 4) & 15, kx = kwin & 15;

  __shared__ alignas(16) _Float16 qs[64][34];
  __shared__ alignas(16) _Float16 ksm[64][34];
  __shared__ alignas(16) _Float16 vs[32][66];
  __shared__ alignas(16) _Float16 attnH[64][66];
  __shared__ float attnF[64][64];
  __shared__ float qn[64], kn[64];
  __shared__ float rpb_s[225];

  int tid = threadIdx.x, lane = tid & 31, wave = tid >> 5;

  for (int i = tid; i < 225; i += 128) rpb_s[i] = rpb[i * 6 + head];

  for (int e = tid; e < 2048; e += 128) {
    int t = e >> 5, c = e & 31;
    int ty = t >> 3, tx = t & 7;
    {
      int y = (wy * 8 + ty + 4) & 127, x = (wx * 8 + tx + 4) & 127;
      qs[t][c] = (_Float16)Qc[(((long)(Bp * 192 + head * 32 + c)) << 14) + (y << 7) + x];
    }
    {
      int y = (ky * 8 + ty + 4) & 127, x = (kx * 8 + tx + 4) & 127;
      long base = ((long)(kb * 384 + head * 32 + c)) << 14;
      ksm[t][c] = (_Float16)KVd[base + (y << 7) + x];
      vs[c][t]  = (_Float16)KVd[base + (192L << 14) + (y << 7) + x];
    }
  }
  __syncthreads();

  if (tid < 64) {
    float s = 0.0f;
#pragma unroll
    for (int c = 0; c < 32; ++c) { float v = (float)qs[tid][c]; s += v * v; }
    qn[tid] = 1.0f / fmaxf(sqrtf(s), 1e-12f);
  } else {
    int t = tid - 64;
    float s = 0.0f;
#pragma unroll
    for (int c = 0; c < 32; ++c) { float v = (float)ksm[t][c]; s += v * v; }
    kn[t] = 1.0f / fmaxf(sqrtf(s), 1e-12f);
  }
  __syncthreads();

  float tmp = temp1[head];
  int mj = (win & 1023) >> 2;       // mask base window (16x16 grid @ 128x128)
  int by = mj >> 4, bx = mj & 15;

  { // logits: wave computes rows [wave*16, wave*16+16), all 64 columns
    v16h a = frag_a(&qs[wave * 16 + (lane & 15)][0], lane);
#pragma unroll
    for (int mt = 0; mt < 4; ++mt) {
      v16h bf = frag_b(&ksm[mt * 16 + (lane & 15)][0], lane);
      v8f c8 = v8f{};
      c8 = __builtin_amdgcn_wmma_f32_16x16x32_f16(
          false, a, false, bf, (short)0, c8, false, false);
      int col = mt * 16 + (lane & 15);
      int cy = col >> 3, cx = col & 7;
      int CY = by * 8 + cy, CX = bx * 8 + cx;
      int regC = ((CY < 120) ? 0 : (CY < 124 ? 1 : 2)) * 3
               + ((CX < 120) ? 0 : (CX < 124 ? 1 : 2));
      float kni = kn[col];
#pragma unroll
      for (int r = 0; r < 8; ++r) {
        int row = wave * 16 + r + ((lane < 16) ? 0 : 8);
        int ry = row >> 3, rx = row & 7;
        int RY = by * 8 + ry, RX = bx * 8 + rx;
        int regR = ((RY < 120) ? 0 : (RY < 124 ? 1 : 2)) * 3
                 + ((RX < 120) ? 0 : (RX < 124 ? 1 : 2));
        float val = c8[r] * tmp * qn[row] * kni
                  + rpb_s[(ry - cy + 7) * 15 + (rx - cx + 7)]
                  + ((regR != regC) ? -100.0f : 0.0f);
        attnF[row][col] = val;
      }
    }
  }
  __syncthreads();

  if (tid < 64) {
    float mx = -1e30f;
    for (int m = 0; m < 64; ++m) mx = fmaxf(mx, attnF[tid][m]);
    float sum = 0.0f;
    for (int m = 0; m < 64; ++m) {
      float v = __expf(attnF[tid][m] - mx);
      attnF[tid][m] = v;
      sum += v;
    }
    float inv = 1.0f / sum;
    for (int m = 0; m < 64; ++m)
      attnH[tid][m] = (_Float16)(attnF[tid][m] * inv);
  }
  __syncthreads();

  // out[n][c] = sum_m attn[n][m] * v[c][m]; K=64 -> 2 WMMA steps per tile
#pragma unroll
  for (int ct = 0; ct < 2; ++ct) {
    v8f acc = v8f{};
#pragma unroll
    for (int kstep = 0; kstep < 64; kstep += 32) {
      v16h a2 = frag_a(&attnH[wave * 16 + (lane & 15)][kstep], lane);
      v16h bf = frag_b(&vs[ct * 16 + (lane & 15)][kstep], lane);
      acc = __builtin_amdgcn_wmma_f32_16x16x32_f16(
          false, a2, false, bf, (short)0, acc, false, false);
    }
    int chan = head * 32 + ct * 16 + (lane & 15);
    long tokBase = (long)win * 64 + wave * 16 + ((lane < 16) ? 0 : 8);
#pragma unroll
    for (int r = 0; r < 8; ++r)
      X1[(long)chan * NT1 + tokBase + r] = acc[r];
  }
}

// Final: out = out0proj + window_reverse(roll(+4)) of X1p
__global__ __launch_bounds__(256) void k_combine(
    const float* __restrict__ C0, const float* __restrict__ X1p,
    float* __restrict__ out, long total)
{
  const long NT1 = 262144;
  long idx = (long)blockIdx.x * 256 + threadIdx.x;
  if (idx >= total) return;
  int b = (int)(idx >> 24);                 // / (256*65536)
  long rem = idx & ((1L << 24) - 1);
  int o = (int)(rem >> 16);
  int p = (int)(rem & 65535);
  int y = p >> 8, x = p & 255;
  int y0 = (y - 4) & 255, x0 = (x - 4) & 255;
  int j = b * 1024 + (y0 >> 3) * 32 + (x0 >> 3);
  int tok = (y0 & 7) * 8 + (x0 & 7);
  out[idx] = C0[idx] + X1p[(long)o * NT1 + (long)j * 64 + tok];
}

// ---------------------------------------------------------------------------
extern "C" void kernel_launch(void* const* d_in, const int* in_sizes, int n_in,
                              void* d_out, int out_size, void* d_ws, size_t ws_size,
                              hipStream_t stream) {
  (void)in_sizes; (void)n_in; (void)out_size; (void)ws_size;
  const float* x      = (const float*)d_in[0];
  const float* Wq0    = (const float*)d_in[1];
  const float* Wqdw0  = (const float*)d_in[2];
  const float* Wkv0   = (const float*)d_in[3];
  const float* Wkvdw0 = (const float*)d_in[4];
  const float* Wq1    = (const float*)d_in[5];
  const float* Wqdw1  = (const float*)d_in[6];
  const float* Wkv1   = (const float*)d_in[7];
  const float* Wkvdw1 = (const float*)d_in[8];
  const float* Wproj0 = (const float*)d_in[9];
  const float* Wproj1 = (const float*)d_in[10];
  const float* temp0  = (const float*)d_in[11];
  const float* temp1  = (const float*)d_in[12];
  const float* rpbT   = (const float*)d_in[13];
  const float* Wds    = (const float*)d_in[14];
  float* out = (float*)d_out;

  const long HW  = 65536;
  const long SA  = 4L * 192 * HW;        // 50,331,648 floats
  const long NB2 = 4L * 192 * 16384;     // 12,582,912 floats
  float* ws   = (float*)d_ws;            // needs ~1.21 GB
  float* bufA = ws;                      // SA
  float* bufB = ws + SA;                 // SA
  float* bufC = ws + 2 * SA;             // 2*SA
  float* bufD = ws + 4 * SA;             // 2*SA
  float* S0   = ws + 6 * SA;             // 24576
  float* qn0  = S0 + 24576;              // 768
  float* kn0  = qn0 + 768;               // 768
  _Float16* A0h = (_Float16*)(kn0 + 768);// 24576 halves
  float* P1 = bufD;                      // pooled   (NB2)
  float* P2 = bufD + NB2;                // xds      (NB2)
  float* K1 = bufD + 2 * NB2;            // kv1      (2*NB2)
  float* K2 = bufD + 4 * NB2;            // kv1 dw   (2*NB2)
  float* X1p = bufD;                     // reused after attn1

  // ------------------ level 0 ------------------
  k_gemm<<<dim3(512, 3, 4), 256, 0, stream>>>(x, Wq0, nullptr, bufA,
      192, 192, (int)HW, 192 * HW, 192 * HW);
  k_dwconv3<<<(unsigned)((4L * 192 * HW + 255) / 256), 256, 0, stream>>>(
      bufA, Wqdw0, bufB, 192, 256, 256, 4L * 192 * HW);
  k_gemm<<<dim3(512, 6, 4), 256, 0, stream>>>(x, Wkv0, nullptr, bufC,
      384, 192, (int)HW, 192 * HW, 384 * HW);
  k_dwconv3<<<(unsigned)((4L * 384 * HW + 255) / 256), 256, 0, stream>>>(
      bufC, Wkvdw0, bufD, 384, 256, 256, 4L * 384 * HW);
  k_rownorm<<<dim3(192, 4), 256, 0, stream>>>(bufB, qn0, HW, 192 * HW, (int)HW, 192);
  k_rownorm<<<dim3(192, 4), 256, 0, stream>>>(bufD, kn0, HW, 384 * HW, (int)HW, 192);
  k_zero<<<96, 256, 0, stream>>>(S0, 24576);
  k_s0_partial<<<dim3(256, 1, 24), 128, 0, stream>>>(bufB, bufD, S0);
  k_s0_softmax<<<24, 32, 0, stream>>>(S0, qn0, kn0, temp0, A0h);
  k_attn0_av<<<dim3(512, 1, 24), 128, 0, stream>>>(A0h, bufD, bufA);
  k_gemm<<<dim3(512, 3, 4), 256, 0, stream>>>(bufA, Wproj0, nullptr, bufC,
      192, 192, (int)HW, 192 * HW, 192 * HW);   // out0proj -> bufC

  // ------------------ level 1 ------------------
  k_gemm<<<dim3(512, 3, 4), 256, 0, stream>>>(x, Wq1, nullptr, bufA,
      192, 192, (int)HW, 192 * HW, 192 * HW);
  k_permute_q1<<<(unsigned)((16L * 192 * 16384 + 255) / 256), 256, 0, stream>>>(
      bufA, bufB, 16L * 192 * 16384);
  k_dwconv3<<<(unsigned)((16L * 192 * 16384 + 255) / 256), 256, 0, stream>>>(
      bufB, Wqdw1, bufA, 192, 128, 128, 16L * 192 * 16384);   // Qc -> bufA
  k_avgpool2<<<(unsigned)((NB2 + 255) / 256), 256, 0, stream>>>(x, P1, NB2);
  k_gemm<<<dim3(128, 3, 4), 256, 0, stream>>>(P1, Wds, P1, P2,
      192, 192, 16384, 192L * 16384, 192L * 16384);           // xds
  k_gemm<<<dim3(128, 6, 4), 256, 0, stream>>>(P2, Wkv1, nullptr, K1,
      384, 192, 16384, 192L * 16384, 384L * 16384);
  k_dwconv3<<<(unsigned)((4L * 384 * 16384 + 255) / 256), 256, 0, stream>>>(
      K1, Wkvdw1, K2, 384, 128, 128, 4L * 384 * 16384);
  k_attn1<<<dim3(4096, 6), 128, 0, stream>>>(bufA, K2, rpbT, temp1, bufB); // X1 -> bufB
  k_gemm<<<dim3(2048, 3, 1), 256, 0, stream>>>(bufB, Wproj1, nullptr, X1p,
      192, 192, 262144, 0, 0);
  k_combine<<<(unsigned)((4L * 192 * HW + 255) / 256), 256, 0, stream>>>(
      bufC, X1p, out, 4L * 192 * HW);
}
